// MultiHeadAttention_5059471474750
// MI455X (gfx1250) — compile-verified, matrix operand order checked by reading, not run on hardware
//
#include <hip/hip_runtime.h>

// MI455X / gfx1250 multi-head causal attention, all matmuls on v_wmma_f32_16x16x32_bf16.
// K-tile staging via Tensor Data Mover (tensor_load_to_lds + s_wait_tensorcnt).
// B=4 T=2048 E=1024 H=16 D=64.

constexpr int B = 4, T = 2048, E = 1024, H = 16, D = 64;

typedef __attribute__((ext_vector_type(16))) __bf16 bf16x16;
typedef __attribute__((ext_vector_type(2)))  __bf16 bf16x2;
typedef __attribute__((ext_vector_type(8)))  float  f32x8;
typedef __attribute__((ext_vector_type(4)))  unsigned u32x4;
typedef __attribute__((ext_vector_type(8)))  unsigned u32x8;

union FragB16 { uint4 u[2]; bf16x16 v; };

#define DEVFN static __device__ __forceinline__

// native converts -> v_cvt_pk_bf16_f32 instead of manual RNE bit twiddling
DEVFN unsigned short f2bf(float f) {
  __bf16 h = (__bf16)f;
  return __builtin_bit_cast(unsigned short, h);
}

DEVFN unsigned pack2(float a, float b) {
  bf16x2 v = {(__bf16)a, (__bf16)b};
  return __builtin_bit_cast(unsigned, v);
}

DEVFN uint4 cvt8(float4 a, float4 b) {
  uint4 r;
  r.x = pack2(a.x, a.y); r.y = pack2(a.z, a.w);
  r.z = pack2(b.x, b.y); r.w = pack2(b.z, b.w);
  return r;
}

// wave32: reduce within each 16-lane half (rows of the C-tile live per-half)
DEVFN float rmax16(float x) {
#pragma unroll
  for (int m = 1; m < 16; m <<= 1) x = fmaxf(x, __shfl_xor(x, m, 32));
  return x;
}
DEVFN float rsum16(float x) {
#pragma unroll
  for (int m = 1; m < 16; m <<= 1) x += __shfl_xor(x, m, 32);
  return x;
}

DEVFN f32x8 wmma_bf16(bf16x16 a, bf16x16 b, f32x8 c) {
  // D = A(16x32 bf16) * B(32x16 bf16) + C(16x16 f32)
  return __builtin_amdgcn_wmma_f32_16x16x32_bf16(false, a, false, b, (short)0, c,
                                                 false, false);
}

// ---- Tensor Data Mover: 2D bf16 tile (rows x cols), row stride in elements ----
// Builds D# group0/group1 per CDNA5 ISA and issues
// tensor_load_to_lds (2-group form: tensors up to 2D). Uniform (SGPR) operands.
DEVFN void tdm_load_tile_2d(unsigned ldsByteOff, const void* gptr,
                            unsigned cols, unsigned rows,
                            unsigned rowStrideElems, unsigned tensorRows) {
  unsigned long long ga = (unsigned long long)(uintptr_t)gptr;
  u32x4 g0;
  g0[0] = 1u;                                       // count=1, user mode
  g0[1] = ldsByteOff;                               // lds_addr (bytes)
  g0[2] = (unsigned)(ga & 0xffffffffu);             // global_addr[31:0]
  g0[3] = (unsigned)((ga >> 32) & 0x01ffffffu)      // global_addr[56:32]
          | 0x80000000u;                            // type=2 ("image")
  u32x8 g1;
  g1[0] = 0x00010000u;                              // data_size=1 (2 bytes)
  g1[1] = (cols & 0xffffu) << 16;                   // tensor_dim0[15:0]
  g1[2] = ((cols >> 16) & 0xffffu) | ((tensorRows & 0xffffu) << 16);  // dim0 hi, dim1 lo
  g1[3] = ((tensorRows >> 16) & 0xffffu) | ((cols & 0xffffu) << 16);  // dim1 hi, tile_dim0
  g1[4] = rows & 0xffffu;                           // tile_dim1 (tile_dim2=0)
  g1[5] = rowStrideElems;                           // tensor_dim0_stride[31:0]
  g1[6] = 0u;                                       // stride hi, dim1_stride lo
  g1[7] = 0u;
  asm volatile("tensor_load_to_lds %0, %1" :: "s"(g0), "s"(g1) : "memory");
}

// ---------------- weight prep ----------------
// W [H,E,D] fp32 -> Wt [H,D,E] bf16 (so B-fragment reads are contiguous per lane)
__global__ void __launch_bounds__(256) wtrans_kernel(const float* __restrict__ W,
                                                     unsigned short* __restrict__ Wt) {
  size_t i = (size_t)blockIdx.x * 256 + threadIdx.x;
  if (i >= (size_t)H * E * D) return;
  int d = (int)(i % D);
  size_t t = i / D;
  int e = (int)(t % E);
  int h = (int)(t / E);
  Wt[((size_t)h * D + d) * E + e] = f2bf(W[i]);
}

__global__ void __launch_bounds__(256) wconv_kernel(const float* __restrict__ W,
                                                    unsigned short* __restrict__ Wt,
                                                    int n) {
  int i = blockIdx.x * 256 + threadIdx.x;
  if (i < n) Wt[i] = f2bf(W[i]);
}

// ---------------- per-head input projection ----------------
// X [B,T,E] fp32, Wt [H,D,E] bf16 -> Out [B,H,T,D] bf16
__global__ void __launch_bounds__(128) proj_kernel(const float* __restrict__ X,
                                                   const unsigned short* __restrict__ Wt,
                                                   unsigned short* __restrict__ Out) {
  const int lane = threadIdx.x & 31;
  const int wave = threadIdx.x >> 5;
  const int l15 = lane & 15, hi = lane >> 4;
  const int t0 = blockIdx.x * 16;
  const int h = blockIdx.y;
  const int b = blockIdx.z;
  const int n0 = wave * 16;

  const float* xrow = X + ((size_t)b * T + (t0 + l15)) * E;
  const unsigned short* wrow = Wt + ((size_t)h * D + (n0 + l15)) * E;

  f32x8 acc = {};
  for (int k0 = 0; k0 < E; k0 += 32) {
    FragB16 af, wf;
    const float4* xa = (const float4*)(xrow + k0 + 8 * hi);
    const float4* xb = (const float4*)(xrow + k0 + 16 + 8 * hi);
    af.u[0] = cvt8(xa[0], xa[1]);
    af.u[1] = cvt8(xb[0], xb[1]);
    wf.u[0] = *(const uint4*)(wrow + k0 + 8 * hi);
    wf.u[1] = *(const uint4*)(wrow + k0 + 16 + 8 * hi);
    acc = wmma_bf16(af.v, wf.v, acc);
  }
  unsigned short* obase = Out + (size_t)(b * H + h) * T * D;
#pragma unroll
  for (int v = 0; v < 8; ++v) {
    int t = t0 + v + 8 * hi;
    obase[(size_t)t * D + n0 + l15] = f2bf(acc[v]);
  }
}

// ---------------- flash attention (causal) ----------------
// grid (T/128, H, B), block = 256 = 8 waves; wave w owns q-rows [qBase+16w, +16)
__global__ void __launch_bounds__(256) attn_kernel(const unsigned short* __restrict__ Qp,
                                                   const unsigned short* __restrict__ Kp,
                                                   const unsigned short* __restrict__ Vp,
                                                   unsigned short* __restrict__ Op) {
  __shared__ unsigned short Kt[32 * 64];     // [s][d] row-major, filled by TDM
  __shared__ unsigned short Vt[64 * 32];     // [d][s] transposed for B-fragments
  __shared__ unsigned short Pb[8][16 * 32];  // wave-private P staging

  const int tid = threadIdx.x;
  const int lane = tid & 31, wave = tid >> 5;
  const int l15 = lane & 15, hi = lane >> 4;
  const int h = blockIdx.y, b = blockIdx.z;
  const int bh = b * H + h;
  const int qBase = blockIdx.x * 128;
  const int q0 = qBase + wave * 16;

  // persistent Q fragments (16 rows x 64 d = 2 A-fragments)
  FragB16 qf[2];
  {
    const unsigned short* qrow = Qp + ((size_t)bh * T + (q0 + l15)) * D;
#pragma unroll
    for (int dc = 0; dc < 2; ++dc) {
      qf[dc].u[0] = *(const uint4*)(qrow + dc * 32 + 8 * hi);
      qf[dc].u[1] = *(const uint4*)(qrow + dc * 32 + 16 + 8 * hi);
    }
  }

  f32x8 o[4] = {};
  float m[8], l[8];
#pragma unroll
  for (int v = 0; v < 8; ++v) { m[v] = -3.0e38f; l[v] = 0.0f; }

  const int sEnd = qBase + 128;  // causal: last chunk any wave in this block needs
  const unsigned short* kbase = Kp + (size_t)bh * T * D;
  const unsigned short* vbase = Vp + (size_t)bh * T * D;
  const unsigned ldsOffK = (unsigned)(unsigned long long)(uintptr_t)&Kt[0];

  for (int s0 = 0; s0 < sEnd; s0 += 32) {
    // --- K tile 32x64 bf16: Tensor Data Mover, one descriptor per workgroup ---
    if (wave == 0) {
      tdm_load_tile_2d(ldsOffK, kbase + (size_t)s0 * D,
                       /*cols=*/D, /*rows=*/32, /*rowStride=*/D, /*tensorRows=*/T);
    }
    // --- V tile 32x64 -> transposed [d][s] in LDS (cooperative VMEM path) ---
    {
      int row = tid >> 3, cg = tid & 7;
      uint4 vq = *(const uint4*)(vbase + (size_t)(s0 + row) * D + cg * 8);
      unsigned short ve[8] = {
          (unsigned short)(vq.x), (unsigned short)(vq.x >> 16),
          (unsigned short)(vq.y), (unsigned short)(vq.y >> 16),
          (unsigned short)(vq.z), (unsigned short)(vq.z >> 16),
          (unsigned short)(vq.w), (unsigned short)(vq.w >> 16)};
#pragma unroll
      for (int j = 0; j < 8; ++j) Vt[(cg * 8 + j) * 32 + row] = ve[j];
      if (s0 + 32 < sEnd) {  // global_prefetch_b8 of next V chunk
        __builtin_prefetch(vbase + (size_t)(s0 + 32 + row) * D + cg * 8, 0, 1);
      }
    }
    if (wave == 0) __builtin_amdgcn_s_wait_tensorcnt(0);  // TDM done before barrier
    __syncthreads();

    if (s0 < q0 + 16) {  // wave-uniform causal activity test (EXEC stays full)
      // ---- scores: two 16x16 tiles over this 32-key chunk ----
      f32x8 sc[2];
#pragma unroll
      for (int st = 0; st < 2; ++st) {
        f32x8 acc = {};
        int r = st * 16 + l15;  // key row in LDS
#pragma unroll
        for (int dc = 0; dc < 2; ++dc) {
          FragB16 kf;
          kf.u[0] = *(const uint4*)(&Kt[r * 64 + dc * 32 + 8 * hi]);
          kf.u[1] = *(const uint4*)(&Kt[r * 64 + dc * 32 + 16 + 8 * hi]);
          acc = wmma_bf16(qf[dc].v, kf.v, acc);
        }
        sc[st] = acc;
      }
      // ---- online softmax over the 32 columns ----
      const float scale = 0.03125f;  // 1/sqrt(E) = 1/32 (reference scales by sqrt(E))
#pragma unroll
      for (int v = 0; v < 8; ++v) {
        int trow = q0 + v + 8 * hi;
        float a0 = sc[0][v] * scale;
        float a1 = sc[1][v] * scale;
        if (s0 + l15 > trow)      a0 = -1.0e30f;  // causal mask (per-lane select)
        if (s0 + 16 + l15 > trow) a1 = -1.0e30f;
        float tm = rmax16(fmaxf(a0, a1));
        float nm = fmaxf(m[v], tm);
        float corr = __expf(m[v] - nm);
        float p0 = __expf(a0 - nm);
        float p1 = __expf(a1 - nm);
        float rs = rsum16(p0 + p1);
        l[v] = l[v] * corr + rs;
        m[v] = nm;
#pragma unroll
        for (int nt = 0; nt < 4; ++nt) o[nt][v] *= corr;
        Pb[wave][(v + 8 * hi) * 32 + l15] = f2bf(p0);
        Pb[wave][(v + 8 * hi) * 32 + 16 + l15] = f2bf(p1);
      }
      __threadfence_block();  // order same-wave P store -> A-fragment reload

      // ---- P (16x32) x V (32x64): 4 WMMAs into O accumulators ----
      FragB16 pf;
      pf.u[0] = *(const uint4*)(&Pb[wave][l15 * 32 + 8 * hi]);
      pf.u[1] = *(const uint4*)(&Pb[wave][l15 * 32 + 16 + 8 * hi]);
#pragma unroll
      for (int nt = 0; nt < 4; ++nt) {
        int d = nt * 16 + l15;
        FragB16 vf;
        vf.u[0] = *(const uint4*)(&Vt[d * 32 + 8 * hi]);
        vf.u[1] = *(const uint4*)(&Vt[d * 32 + 16 + 8 * hi]);
        o[nt] = wmma_bf16(pf.v, vf.v, o[nt]);
      }
    }
    __syncthreads();
  }

  unsigned short* obase = Op + (size_t)bh * T * D;
#pragma unroll
  for (int v = 0; v < 8; ++v) {
    float inv = 1.0f / l[v];
    int t = q0 + v + 8 * hi;
#pragma unroll
    for (int nt = 0; nt < 4; ++nt) {
      obase[(size_t)t * D + nt * 16 + l15] = f2bf(o[nt][v] * inv);
    }
  }
}

// ---------------- output projection ----------------
// Ac [B,H,T,D] bf16 viewed as concat-heads (B*T, E); out = Ac @ Wp^T + bp
__global__ void __launch_bounds__(128) outproj_kernel(const unsigned short* __restrict__ Ac,
                                                      const unsigned short* __restrict__ Wpb,
                                                      const float* __restrict__ bp,
                                                      float* __restrict__ Out) {
  const int lane = threadIdx.x & 31, wave = threadIdx.x >> 5;
  const int l15 = lane & 15, hi = lane >> 4;
  const int m0 = blockIdx.x * 16;
  const int n0 = blockIdx.y * 64 + wave * 16;

  const int r = m0 + l15;  // A-fragment row
  const int bb = r / T, tt = r % T;
  const unsigned short* wrow = Wpb + (size_t)(n0 + l15) * E;  // Wp[j, :] (B = Wp^T)

  f32x8 acc = {};
  for (int k0 = 0; k0 < E; k0 += 32) {
    const int hh = k0 >> 6;  // head containing this K-chunk (D=64, k0 % 32 == 0)
    const int dl = k0 & 63;
    const unsigned short* arow = Ac + ((size_t)(bb * H + hh) * T + tt) * D;
    FragB16 af, wf;
    af.u[0] = *(const uint4*)(arow + dl + 8 * hi);
    af.u[1] = *(const uint4*)(arow + dl + 16 + 8 * hi);
    wf.u[0] = *(const uint4*)(wrow + k0 + 8 * hi);
    wf.u[1] = *(const uint4*)(wrow + k0 + 16 + 8 * hi);
    acc = wmma_bf16(af.v, wf.v, acc);
  }
  const float bias = bp[n0 + l15];
#pragma unroll
  for (int v = 0; v < 8; ++v) {
    int rr = m0 + v + 8 * hi;
    Out[(size_t)rr * E + n0 + l15] = acc[v] + bias;
  }
}

// ---------------- launcher ----------------
extern "C" void kernel_launch(void* const* d_in, const int* in_sizes, int n_in,
                              void* d_out, int out_size, void* d_ws, size_t ws_size,
                              hipStream_t stream) {
  (void)in_sizes; (void)n_in; (void)out_size; (void)ws_size;
  const float* k_in = (const float*)d_in[0];
  const float* q_in = (const float*)d_in[1];
  const float* v_in = (const float*)d_in[2];
  /* d_in[3] = mask: causal triu(k=1), structure is known -> unused */
  const float* Wk = (const float*)d_in[4];
  const float* Wq = (const float*)d_in[5];
  const float* Wv = (const float*)d_in[6];
  const float* Wp = (const float*)d_in[7];
  const float* bp = (const float*)d_in[8];
  float* out = (float*)d_out;

  char* ws = (char*)d_ws;
  const size_t szW = (size_t)H * E * D * sizeof(unsigned short);      // 2 MB
  const size_t szP = (size_t)B * H * T * D * sizeof(unsigned short);  // 16 MB
  unsigned short* WqT = (unsigned short*)ws; ws += szW;
  unsigned short* WkT = (unsigned short*)ws; ws += szW;
  unsigned short* WvT = (unsigned short*)ws; ws += szW;
  unsigned short* WpB = (unsigned short*)ws; ws += (size_t)E * E * sizeof(unsigned short);
  unsigned short* Qp = (unsigned short*)ws; ws += szP;
  unsigned short* Kp = (unsigned short*)ws; ws += szP;
  unsigned short* Vp = (unsigned short*)ws; ws += szP;
  unsigned short* Ao = (unsigned short*)ws; ws += szP;  // total ~72 MB

  const int wElems = H * E * D;
  wtrans_kernel<<<(wElems + 255) / 256, 256, 0, stream>>>(Wq, WqT);
  wtrans_kernel<<<(wElems + 255) / 256, 256, 0, stream>>>(Wk, WkT);
  wtrans_kernel<<<(wElems + 255) / 256, 256, 0, stream>>>(Wv, WvT);
  wconv_kernel<<<(E * E + 255) / 256, 256, 0, stream>>>(Wp, WpB, E * E);

  dim3 pg(T / 16, H, B);
  proj_kernel<<<pg, 128, 0, stream>>>(q_in, WqT, Qp);
  proj_kernel<<<pg, 128, 0, stream>>>(k_in, WkT, Kp);
  proj_kernel<<<pg, 128, 0, stream>>>(v_in, WvT, Vp);

  attn_kernel<<<dim3(T / 128, H, B), 256, 0, stream>>>(Qp, Kp, Vp, Ao);

  outproj_kernel<<<dim3((B * T) / 16, E / 64), 128, 0, stream>>>(Ao, WpB, bp, out);
}
